// StaticCSRRouter_20813411516527
// MI455X (gfx1250) — matrix operand order repaired and checked by reading, not compiled, not run on hardware
//
#include <hip/hip_runtime.h>
#include <stdint.h>

#define B_ROWS   8192
#define IN_DIM   4096
#define OUT_DIM  4096   // 8 experts * 512
#define NEXP     8

#define BM 128
#define BN 128
#define BK 64
#define LDS_ROW_B 144                 // (64 + 8 pad) bf16 elems * 2 bytes
#define TILE_BYTES (128 * LDS_ROW_B)  // 18432
#define BUF_BYTES  (2 * TILE_BYTES)   // 36864 (X tile + W tile)

typedef __attribute__((ext_vector_type(16))) __bf16 v16bf;
typedef __attribute__((ext_vector_type(8)))  float  v8f;
typedef __attribute__((ext_vector_type(4)))  unsigned int v4u;
typedef __attribute__((ext_vector_type(8)))  int v8i;
typedef __attribute__((ext_vector_type(4)))  int v4i;

union Frag { v16bf v; uint4 q[2]; };

// ---------------- fp32 -> bf16 pre-pass (round-to-nearest) ----------------
__device__ inline unsigned int pack_bf16x2(float a, float b) {
  unsigned int ua = __float_as_uint(a);
  unsigned int ub = __float_as_uint(b);
  ua += 0x7FFFu + ((ua >> 16) & 1u);
  ub += 0x7FFFu + ((ub >> 16) & 1u);
  return (ua >> 16) | (ub & 0xFFFF0000u);
}

__global__ __launch_bounds__(256) void cvt_f32_to_bf16(
    const float4* __restrict__ src, uint2* __restrict__ dst, int n4) {
  int i = blockIdx.x * blockDim.x + threadIdx.x;
  if (i < n4) {
    float4 f = src[i];
    dst[i] = make_uint2(pack_bf16x2(f.x, f.y), pack_bf16x2(f.z, f.w));
  }
}

// ---------------- TDM: 2D bf16 tile (64 K-elems x 128 rows) -> LDS --------
// Adds 16B LDS padding per 128B row: pad_interval code 4 (32 DW), pad_amount
// code 3 (4 DW) -> LDS row stride 144B.
__device__ inline void tdm_load_tile(unsigned int lds_addr, const void* gptr,
                                     unsigned int tensor_rows) {
  uint64_t ga = (uint64_t)(uintptr_t)gptr;
  v4u g0 = {
      1u,                                                   // count=1 (valid)
      lds_addr,                                             // LDS byte address
      (unsigned int)ga,                                     // global addr lo
      ((unsigned int)(ga >> 32) & 0x1FFFFFFu) | (2u << 30)  // addr hi | type=2
  };
  v8i g1 = {
      (int)((1u << 16) | (1u << 20) | (4u << 22) | (3u << 25)), // 2B elems, pad
      (int)((IN_DIM & 0xFFFFu) << 16),                          // tensor_dim0 lo
      (int)((IN_DIM >> 16) | ((tensor_rows & 0xFFFFu) << 16)),  // d0 hi | d1 lo
      (int)((tensor_rows >> 16) | (64u << 16)),                 // d1 hi | tile_dim0=64
      (int)128,                                                 // tile_dim1=128, tile_dim2=0
      (int)IN_DIM,                                              // dim0 stride lo32
      0, 0
  };
  v4i z = {0, 0, 0, 0};
#if __clang_major__ >= 23
  v8i z8 = {0, 0, 0, 0, 0, 0, 0, 0};
  __builtin_amdgcn_tensor_load_to_lds(g0, g1, z, z, z8, 0);
#else
  __builtin_amdgcn_tensor_load_to_lds(g0, g1, z, z, 0);
#endif
}

// ---------------- bf16 WMMA GEMM + tanh epilogue --------------------------
__global__ __launch_bounds__(256) void moe_gemm_tanh(
    const unsigned short* __restrict__ xb,   // bf16 X  [8192][4096]
    const unsigned short* __restrict__ wb,   // bf16 W2 [4096][4096] (N-major)
    float* __restrict__ out)                 // [8192][4096]
{
  __shared__ __align__(16) unsigned char sh[2][BUF_BYTES];  // 72 KB, dbl-buffered

  const int tid   = threadIdx.x;
  const int lane  = tid & 31;
  const int wid   = tid >> 5;
  const int wm    = wid & 3;    // 4 waves tile M (32 rows each)
  const int wn    = wid >> 2;   // 2 waves tile N (64 cols each)
  const int mlane = lane & 15;
  const int hi    = lane >> 4;

  const unsigned int lds_base = (unsigned int)(uintptr_t)&sh[0][0];
  const uint64_t xg = (uint64_t)(uintptr_t)(xb + (size_t)blockIdx.y * BM * IN_DIM);
  const uint64_t wg = (uint64_t)(uintptr_t)(wb + (size_t)blockIdx.x * BN * IN_DIM);

  v8f acc[2][4] = {};

  const bool issuer = (wid == 0);
  const int NK = IN_DIM / BK;  // 64

  if (issuer) {
    tdm_load_tile(lds_base, (const void*)(uintptr_t)xg, B_ROWS);
    tdm_load_tile(lds_base + TILE_BYTES, (const void*)(uintptr_t)wg, OUT_DIM);
  }

  for (int it = 0; it < NK; ++it) {
    const int buf = it & 1;
    if (issuer) {
      if (it + 1 < NK) {
        const uint64_t koff = (uint64_t)(it + 1) * (BK * 2);  // bytes along K
        const unsigned int lb = lds_base + (unsigned int)((buf ^ 1) * BUF_BYTES);
        tdm_load_tile(lb, (const void*)(uintptr_t)(xg + koff), B_ROWS);
        tdm_load_tile(lb + TILE_BYTES, (const void*)(uintptr_t)(wg + koff), OUT_DIM);
        __builtin_amdgcn_s_wait_tensorcnt(2);  // current buffer's pair done
      } else {
        __builtin_amdgcn_s_wait_tensorcnt(0);
      }
    }
    __syncthreads();

    const unsigned char* xsb = &sh[buf][0];
    const unsigned char* wsb = &sh[buf][TILE_BYTES];

#pragma unroll
    for (int ks = 0; ks < 2; ++ks) {   // two K=32 steps per BK=64
      Frag a[2], b[4];
#pragma unroll
      for (int t = 0; t < 2; ++t) {    // A: lane holds K[hi*8..+8) and +16
        const int row = wm * 32 + t * 16 + mlane;
        const int off = row * LDS_ROW_B + ks * 64 + hi * 16;
        a[t].q[0] = *(const uint4*)(xsb + off);
        a[t].q[1] = *(const uint4*)(xsb + off + 32);
      }
#pragma unroll
      for (int u = 0; u < 4; ++u) {    // B: lane holds K[hi*8..+8) and +16
        const int col = wn * 64 + u * 16 + mlane;
        const int off = col * LDS_ROW_B + ks * 64 + hi * 16;
        b[u].q[0] = *(const uint4*)(wsb + off);
        b[u].q[1] = *(const uint4*)(wsb + off + 32);
      }
#pragma unroll
      for (int t = 0; t < 2; ++t)
#pragma unroll
        for (int u = 0; u < 4; ++u)
          acc[t][u] = __builtin_amdgcn_wmma_f32_16x16x32_bf16(
              false, a[t].v, false, b[u].v, (short)0, acc[t][u], false, false);
    }
    __syncthreads();
  }

  const int row0 = blockIdx.y * BM + wm * 32;
  const int col0 = blockIdx.x * BN + wn * 64;
#pragma unroll
  for (int t = 0; t < 2; ++t)
#pragma unroll
    for (int u = 0; u < 4; ++u) {
      const int cc = col0 + u * 16 + mlane;
#pragma unroll
      for (int r = 0; r < 8; ++r) {
        const int rr = row0 + t * 16 + hi * 8 + r;
        out[(size_t)rr * OUT_DIM + cc] = tanhf(acc[t][u][r]);
      }
    }
}

// ---------------- gating: logits -> softmax (one wave per row) ------------
__global__ __launch_bounds__(256) void gate_softmax(
    const float* __restrict__ x, const float* __restrict__ gw,
    const float* __restrict__ gb, const float* __restrict__ ttm,
    float* __restrict__ wout)
{
  const int lane = threadIdx.x & 31;
  const int wid  = threadIdx.x >> 5;
  const int row  = blockIdx.x * 8 + wid;
  const float* xr = x + (size_t)row * IN_DIM;

  float acc[NEXP];
#pragma unroll
  for (int e = 0; e < NEXP; ++e) acc[e] = 0.f;

  for (int k = lane; k < IN_DIM; k += 32) {
    const float xv = xr[k];
#pragma unroll
    for (int e = 0; e < NEXP; ++e) acc[e] += xv * gw[e * IN_DIM + k];
  }
#pragma unroll
  for (int e = 0; e < NEXP; ++e)
#pragma unroll
    for (int off = 16; off > 0; off >>= 1)
      acc[e] += __shfl_xor(acc[e], off, 32);

  if (lane == 0) {
    float lg[NEXP], mx = -3.402823466e38f;
#pragma unroll
    for (int e = 0; e < NEXP; ++e) { lg[e] = acc[e] + gb[e] + ttm[e]; mx = fmaxf(mx, lg[e]); }
    float s = 0.f;
#pragma unroll
    for (int e = 0; e < NEXP; ++e) { lg[e] = __expf(lg[e] - mx); s += lg[e]; }
    const float inv = 1.f / s;
#pragma unroll
    for (int e = 0; e < NEXP; ++e) wout[(size_t)row * NEXP + e] = lg[e] * inv;
  }
}

// --------------------------------------------------------------------------
extern "C" void kernel_launch(void* const* d_in, const int* in_sizes, int n_in,
                              void* d_out, int out_size, void* d_ws, size_t ws_size,
                              hipStream_t stream) {
  const float* x   = (const float*)d_in[0];
  const float* W   = (const float*)d_in[1];
  const float* gw  = (const float*)d_in[2];
  const float* gb  = (const float*)d_in[3];
  const float* ttm = (const float*)d_in[4];

  float* out     = (float*)d_out;
  float* weights = out + (size_t)B_ROWS * OUT_DIM;

  unsigned short* xb = (unsigned short*)d_ws;                 // 64 MB bf16 X
  unsigned short* wb = xb + (size_t)B_ROWS * IN_DIM;          // 32 MB bf16 W2

  {
    int n4 = (B_ROWS * IN_DIM) / 4;
    cvt_f32_to_bf16<<<(n4 + 255) / 256, 256, 0, stream>>>(
        (const float4*)x, (uint2*)xb, n4);
  }
  {
    int n4 = (OUT_DIM * IN_DIM) / 4;
    cvt_f32_to_bf16<<<(n4 + 255) / 256, 256, 0, stream>>>(
        (const float4*)W, (uint2*)wb, n4);
  }

  dim3 grid(OUT_DIM / BN, B_ROWS / BM);  // (32, 64)
  moe_gemm_tanh<<<grid, 256, 0, stream>>>(xb, wb, out);

  gate_softmax<<<B_ROWS / 8, 256, 0, stream>>>(x, gw, gb, ttm, weights);
}
